// EquivTransAttentionLayer_38500086841986
// MI455X (gfx1250) — matrix operand order repaired and checked by reading, not compiled, not run on hardware
//
#include <hip/hip_runtime.h>
#include <hip/hip_bf16.h>
#include <math.h>

#define NN 50000
#define EE 1600000
#define IN_F 256
#define AD 128
#define NH 8
#define DK 16

typedef __attribute__((ext_vector_type(16))) __bf16 v16bf;
typedef __attribute__((ext_vector_type(8)))  float  v8f;

// fp32 -> bf16: prefer native converts (v_cvt_*bf16_f32) over manual bit ops.
__device__ __forceinline__ __bf16 f2bf(float x) {
#if __has_builtin(__builtin_amdgcn_cvt_pk_bf16_f32)
    // packed convert; take low half
    typedef __attribute__((ext_vector_type(2))) __bf16 v2bf;
    v2bf p = __builtin_amdgcn_cvt_pk_bf16_f32(x, x);
    return p[0];
#else
    return (__bf16)x;   // compiler lowers to native cvt on gfx1250
#endif
}

// order-preserving float<->u32 encoding for atomicMax over signed floats
__device__ __forceinline__ unsigned enc_f32(float f) {
    unsigned u = __builtin_bit_cast(unsigned, f);
    return (u & 0x80000000u) ? ~u : (u | 0x80000000u);
}
__device__ __forceinline__ float dec_f32(unsigned v) {
    unsigned u = (v & 0x80000000u) ? (v ^ 0x80000000u) : ~v;
    return __builtin_bit_cast(float, u);
}

// ---------------------------------------------------------------------------
// Pass 0: zero segment-max (encoded) and segment-sum buffers
// ---------------------------------------------------------------------------
__global__ void __launch_bounds__(256)
init_seg_kernel(unsigned* __restrict__ maxbuf, float* __restrict__ sumbuf, int n) {
    int i = blockIdx.x * 256 + threadIdx.x;
    if (i < n) { maxbuf[i] = 0u; sumbuf[i] = 0.0f; }
}

// ---------------------------------------------------------------------------
// Pass 1: q/k projection GEMM via v_wmma_f32_16x16x32_bf16.
// One wave per 16x16 output tile of qk[N][256] (cols 0..127 = q, 128..255 = k).
// 3125 row tiles x 16 col tiles = 50000 waves, 8 waves/block -> 6250 blocks.
// ---------------------------------------------------------------------------
__global__ void __launch_bounds__(256)
qk_proj_kernel(const float* __restrict__ hi,
               const float* __restrict__ Qw, const float* __restrict__ Qb,
               const float* __restrict__ Kw, const float* __restrict__ Kb,
               float* __restrict__ qk)
{
    const int lane = threadIdx.x & 31;
    const int wave = threadIdx.x >> 5;
    const int tile = blockIdx.x * 8 + wave;     // 0..49999
    const int rowTile = tile >> 4;              // 0..3124 (node tiles)
    const int colTile = tile & 15;              // 0..15   (output-col tiles)
    const int half = lane >> 4;                 // 0 or 1
    const int l15  = lane & 15;

    const int m    = rowTile * 16 + l15;        // A row (node) for this lane
    const int colg = colTile * 16 + l15;        // output column 0..255
    const float* __restrict__ Arow = hi + (long)m * IN_F;
    const float* __restrict__ Wrow = (colg < AD) ? (Qw + (long)colg * IN_F)
                                                 : (Kw + (long)(colg - AD) * IN_F);
    v8f acc = {};
#pragma unroll
    for (int kk = 0; kk < IN_F; kk += 32) {
        // A fragment (16x32 bf16): lane half 0 -> K=kk+0..15,
        //                          lane half 1 -> K=kk+16..31
        const int ka = kk + half * 16;
        float4 a0 = *(const float4*)(Arow + ka);
        float4 a1 = *(const float4*)(Arow + ka + 4);
        float4 a2 = *(const float4*)(Arow + ka + 8);
        float4 a3 = *(const float4*)(Arow + ka + 12);
        v16bf A;
        A[0]=f2bf(a0.x); A[1]=f2bf(a0.y); A[2]=f2bf(a0.z); A[3]=f2bf(a0.w);
        A[4]=f2bf(a1.x); A[5]=f2bf(a1.y); A[6]=f2bf(a1.z); A[7]=f2bf(a1.w);
        A[8]=f2bf(a2.x); A[9]=f2bf(a2.y); A[10]=f2bf(a2.z); A[11]=f2bf(a2.w);
        A[12]=f2bf(a3.x); A[13]=f2bf(a3.y); A[14]=f2bf(a3.z); A[15]=f2bf(a3.w);
        // B fragment (32x16 bf16): B[k][n] = W[n][k]; lane = column, half
        // selects K section kk+0..15 vs kk+16..31, contiguous in W row.
        const int kb = kk + half * 16;
        float4 b0 = *(const float4*)(Wrow + kb);
        float4 b1 = *(const float4*)(Wrow + kb + 4);
        float4 b2 = *(const float4*)(Wrow + kb + 8);
        float4 b3 = *(const float4*)(Wrow + kb + 12);
        v16bf B;
        B[0]=f2bf(b0.x); B[1]=f2bf(b0.y); B[2]=f2bf(b0.z); B[3]=f2bf(b0.w);
        B[4]=f2bf(b1.x); B[5]=f2bf(b1.y); B[6]=f2bf(b1.z); B[7]=f2bf(b1.w);
        B[8]=f2bf(b2.x); B[9]=f2bf(b2.y); B[10]=f2bf(b2.z); B[11]=f2bf(b2.w);
        B[12]=f2bf(b3.x); B[13]=f2bf(b3.y); B[14]=f2bf(b3.z); B[15]=f2bf(b3.w);

        acc = __builtin_amdgcn_wmma_f32_16x16x32_bf16(
            /*neg_a=*/false, A, /*neg_b=*/false, B,
            /*c_mod=*/(short)0, acc, /*reuse_a=*/false, /*reuse_b=*/false);
    }

    const float bias = (colg < AD) ? Qb[colg] : Kb[colg - AD];
    // C/D layout: element v of lane L holds row (v + 8*(L>>4)), col (L&15)
#pragma unroll
    for (int v = 0; v < 8; ++v) {
        int row = rowTile * 16 + v + half * 8;
        qk[(long)row * 256 + colg] = acc[v] + bias;
    }
}

// ---------------------------------------------------------------------------
// Pass 2: per (edge, head) dot product + segment max (encoded u32 atomicMax)
// ---------------------------------------------------------------------------
__global__ void __launch_bounds__(256)
edge_dot_kernel(const float* __restrict__ qk,
                const float* __restrict__ radial,
                const float* __restrict__ Qrw, const float* __restrict__ Qrb,
                const int* __restrict__ edge0, const int* __restrict__ edge1,
                float* __restrict__ prods_out, unsigned* __restrict__ maxbuf)
{
    int idx = blockIdx.x * 256 + threadIdx.x;
    if (idx >= EE * NH) return;
    const int e = idx >> 3, h = idx & 7;
    const int src = edge0[e], dst = edge1[e];
    const float r = radial[e];
    const float* __restrict__ qp = qk + (long)src * 256 + h * DK;        // q half
    const float* __restrict__ kp = qk + (long)dst * 256 + AD + h * DK;   // k half
    const float* __restrict__ rw = Qrw + h * DK;
    const float* __restrict__ rb = Qrb + h * DK;
    float dot = 0.0f;
#pragma unroll
    for (int j = 0; j < DK; j += 4) {
        float4 q4 = *(const float4*)(qp + j);
        float4 k4 = *(const float4*)(kp + j);
        float4 w4 = *(const float4*)(rw + j);
        float4 b4 = *(const float4*)(rb + j);
        dot = fmaf(q4.x + fmaf(r, w4.x, b4.x), k4.x, dot);
        dot = fmaf(q4.y + fmaf(r, w4.y, b4.y), k4.y, dot);
        dot = fmaf(q4.z + fmaf(r, w4.z, b4.z), k4.z, dot);
        dot = fmaf(q4.w + fmaf(r, w4.w, b4.w), k4.w, dot);
    }
    const float p = dot * 0.25f;   // 1/sqrt(DK)
    prods_out[idx] = p;
    atomicMax(&maxbuf[src * NH + h], enc_f32(p));
}

// ---------------------------------------------------------------------------
// Pass 3: ex = exp(p - segmax); write into attention region; atomicAdd sums
// ---------------------------------------------------------------------------
__global__ void __launch_bounds__(256)
edge_exp_kernel(const float* __restrict__ prods,
                const int* __restrict__ edge0,
                const unsigned* __restrict__ maxbuf,
                float* __restrict__ att_out, float* __restrict__ sumbuf)
{
    int idx = blockIdx.x * 256 + threadIdx.x;
    if (idx >= EE * NH) return;
    const int e = idx >> 3, h = idx & 7;
    const int src = edge0[e];
    const float m = dec_f32(maxbuf[src * NH + h]);
    const float ex = __expf(prods[idx] - m);
    att_out[idx] = ex;
    atomicAdd(&sumbuf[src * NH + h], ex);
}

// ---------------------------------------------------------------------------
// Pass 4: normalize by segment sum
// ---------------------------------------------------------------------------
__global__ void __launch_bounds__(256)
edge_norm_kernel(float* __restrict__ att_out,
                 const int* __restrict__ edge0,
                 const float* __restrict__ sumbuf)
{
    int idx = blockIdx.x * 256 + threadIdx.x;
    if (idx >= EE * NH) return;
    const int e = idx >> 3, h = idx & 7;
    const int src = edge0[e];
    att_out[idx] = att_out[idx] / sumbuf[src * NH + h];
}

extern "C" void kernel_launch(void* const* d_in, const int* in_sizes, int n_in,
                              void* d_out, int out_size, void* d_ws, size_t ws_size,
                              hipStream_t stream)
{
    (void)in_sizes; (void)n_in; (void)out_size; (void)ws_size;
    const float* hi     = (const float*)d_in[0];   // [N, 256]
    const float* radial = (const float*)d_in[1];   // [E, 1]
    const float* Qw     = (const float*)d_in[2];   // [128, 256]
    const float* Qb     = (const float*)d_in[3];   // [128]
    const float* Qrw    = (const float*)d_in[4];   // [128, 1]
    const float* Qrb    = (const float*)d_in[5];   // [128]
    const float* Kw     = (const float*)d_in[6];   // [128, 256]
    const float* Kb     = (const float*)d_in[7];   // [128]
    const int*   edge   = (const int*)d_in[8];     // [2, E]
    const int* edge0 = edge;
    const int* edge1 = edge + EE;

    float* att_out   = (float*)d_out;              // [E, 8]
    float* prods_out = (float*)d_out + (long)EE * NH;  // [E, 8]

    // workspace: qk [N][256] f32 | maxbuf [N*8] u32 | sumbuf [N*8] f32
    float*    qk     = (float*)d_ws;
    unsigned* maxbuf = (unsigned*)(qk + (long)NN * 256);
    float*    sumbuf = (float*)(maxbuf + (long)NN * NH);

    const int segN = NN * NH;                      // 400000
    init_seg_kernel<<<(segN + 255) / 256, 256, 0, stream>>>(maxbuf, sumbuf, segN);

    // 50000 tiles, 8 waves (tiles) per 256-thread block
    qk_proj_kernel<<<6250, 256, 0, stream>>>(hi, Qw, Qb, Kw, Kb, qk);

    const long tot = (long)EE * NH;                // 12.8M
    const int eblocks = (int)((tot + 255) / 256);  // 50000
    edge_dot_kernel<<<eblocks, 256, 0, stream>>>(qk, radial, Qrw, Qrb,
                                                 edge0, edge1, prods_out, maxbuf);
    edge_exp_kernel<<<eblocks, 256, 0, stream>>>(prods_out, edge0, maxbuf,
                                                 att_out, sumbuf);
    edge_norm_kernel<<<eblocks, 256, 0, stream>>>(att_out, edge0, sumbuf);
}
